// SpikeDrivenSelfAttention_64553358459233
// MI455X (gfx1250) — matrix operand order repaired and checked by reading, not compiled, not run on hardware
//
#include <hip/hip_runtime.h>
#include <stdint.h>

typedef __attribute__((ext_vector_type(16))) __bf16 v16bf;
typedef __attribute__((ext_vector_type(8)))  float  v8f;

constexpr int C  = 768;
constexpr int B  = 16;
constexpr int N  = 1024;          // H*W
constexpr int BN = B * N;         // 16384
constexpr int WELEM = C * C;      // 589824

// workspace layout (bytes)
constexpr size_t OFF_W   = 0;                               // 4 x [C*C] bf16 weights
constexpr size_t WBYTES  = (size_t)WELEM * 2;               // 1,179,648
constexpr size_t OFF_XS  = OFF_W + 4 * WBYTES;              // bf16 [C, BN]
constexpr size_t XSBYTES = (size_t)C * BN * 2;              // 25,165,824
constexpr size_t OFF_Q   = OFF_XS + XSBYTES;                // bf16 [C, BN]
constexpr size_t OFF_KS  = OFF_Q  + XSBYTES;                // u8   [C, BN]
constexpr size_t OFF_VS  = OFF_KS + (size_t)C * BN;         // u8   [C, BN]
constexpr size_t OFF_KVS = OFF_VS + (size_t)C * BN;         // bf16 [B*C]
constexpr size_t OFF_PAR = OFF_KVS + 2 * (size_t)B * C;     // f32  [8*C] inv/bias x4

__device__ __forceinline__ unsigned short f2bf(float f) {
  union { float f; uint32_t u; } a; a.f = f;
  uint32_t r = a.u + 0x7FFFu + ((a.u >> 16) & 1u);   // round-to-nearest-even
  return (unsigned short)(r >> 16);
}

// ---- fold BatchNorm (+ proj bias) into per-channel inv/bias -----------------
__global__ void prep_params(const float* qg, const float* qb, const float* qm, const float* qv,
                            const float* kg, const float* kb, const float* km, const float* kv,
                            const float* vg, const float* vb, const float* vm, const float* vv,
                            const float* pg, const float* pb, const float* pm, const float* pv,
                            const float* bp, float* P) {
  int c = blockIdx.x * blockDim.x + threadIdx.x;
  if (c >= C) return;
  float i0 = qg[c] * rsqrtf(qv[c] + 1e-5f); P[0*C+c] = i0; P[1*C+c] = qb[c] - qm[c]*i0;
  float i1 = kg[c] * rsqrtf(kv[c] + 1e-5f); P[2*C+c] = i1; P[3*C+c] = kb[c] - km[c]*i1;
  float i2 = vg[c] * rsqrtf(vv[c] + 1e-5f); P[4*C+c] = i2; P[5*C+c] = vb[c] - vm[c]*i2;
  float i3 = pg[c] * rsqrtf(pv[c] + 1e-5f); P[6*C+c] = i3; P[7*C+c] = pb[c] - pm[c]*i3 + bp[c]*i3;
}

// ---- weights fp32 -> bf16 ---------------------------------------------------
__global__ void convert_w(const float* w0, const float* w1, const float* w2, const float* w3,
                          unsigned short* dst) {
  int idx = blockIdx.x * 256 + threadIdx.x;
  if (idx >= 4 * WELEM) return;
  int which = idx / WELEM, r = idx - which * WELEM;
  const float* s = (which == 0) ? w0 : (which == 1) ? w1 : (which == 2) ? w2 : w3;
  dst[idx] = f2bf(s[r]);
}

// ---- shortcut LIF: xs[c, b*N+n] = (x[b,c,n] >= 2) as bf16 0/1 ---------------
__global__ void spike_x(const float* __restrict__ x, unsigned short* __restrict__ xs) {
  int idx = blockIdx.x * 256 + threadIdx.x;       // over B*C*N
  if (idx >= B * C * N) return;
  int n = idx & (N - 1);
  int c = (idx >> 10) % C;
  int b = idx / (C * N);
  xs[(size_t)c * BN + b * N + n] = (x[idx] >= 2.0f) ? 0x3F80 : 0;
}

// ---- WMMA GEMM: D[row, col] = sum_k A[row,k]*Bm[k,col]; fused epilogues -----
// MODE 0: spike -> bf16 out     MODE 1: spike -> u8 out
// MODE 2: B masked by kv spike; out = y*inv+bias + identity, f32
union FragAB { v16bf v; uint4 q[2]; };

template <int MODE>
__global__ __launch_bounds__(256)
void gemm_spike(const unsigned short* __restrict__ A,    // bf16 [C x C] row-major
                const unsigned short* __restrict__ Bm,   // bf16 [C x BN]
                const float* __restrict__ inv,
                const float* __restrict__ bias,
                const unsigned short* __restrict__ kvs,  // bf16 [B*C] (MODE 2)
                const float* __restrict__ xin,           // identity  (MODE 2)
                unsigned short* __restrict__ out16,      // MODE 0
                unsigned char*  __restrict__ out8,       // MODE 1
                float* __restrict__ outf)                // MODE 2
{
  __shared__ __align__(16) unsigned short As[128 * 40];  // row stride 40 halfs (pad 8)
  __shared__ __align__(16) unsigned int   Bs[256 * 20];  // col n stride 20 u32 K-pairs

  const int tid  = threadIdx.x;
  const int lane = tid & 31;
  const int wave = tid >> 5;
  const int wm = wave >> 2, wn = wave & 3;               // 2 x 4 wave grid
  const int m0 = wm * 64, n0 = wn * 64;                  // 64x64 per wave
  const int nl = lane & 15, half = lane >> 4;
  const int mbase  = blockIdx.y * 128;
  const int bnbase = blockIdx.x * 256;
  const int bIdx   = bnbase >> 10;                       // batch (256 | 1024)

  v8f acc[4][4] = {};

  // staging coords
  const int am  = tid >> 1;                 // A: row 0..127
  const int ac0 = (tid & 1) * 16;           // A: col block 0 / 16
  const int pr  = tid >> 4;                 // B: K-pair 0..15
  const int cn  = (tid & 15) * 8;           // B: col block

  unsigned int msk0 = 0xFFFFFFFFu, msk1 = 0xFFFFFFFFu;

  for (int kt = 0; kt < C / 32; ++kt) {
    const int kb = kt * 32;
    __syncthreads();
    // ---- stage A tile (128 x 32 bf16) ----
    {
      const uint4* src = (const uint4*)(A + (size_t)(mbase + am) * C + kb + ac0);
      uint4 v0 = src[0], v1 = src[1];
      *(uint4*)&As[am * 40 + ac0]     = v0;
      *(uint4*)&As[am * 40 + ac0 + 8] = v1;
    }
    // ---- stage B tile (32 x 256 bf16), K-pair packed ----
    if (MODE == 2) {
      msk0 = kvs[bIdx * C + kb + 2 * pr]     ? 0xFFFFFFFFu : 0u;
      msk1 = kvs[bIdx * C + kb + 2 * pr + 1] ? 0xFFFFFFFFu : 0u;
    }
    #pragma unroll
    for (int r = 0; r < 2; ++r) {
      const int nn = cn + r * 128;
      const unsigned short* b0 = Bm + (size_t)(kb + 2 * pr) * BN + bnbase + nn;
      union { uint4 q; unsigned short h[8]; } va, vb;
      va.q = *(const uint4*)b0;
      vb.q = *(const uint4*)(b0 + BN);
      if (MODE == 2) {   // out = q * kv_spike : bf16 {0,1.0} -> bitwise AND
        va.q.x &= msk0; va.q.y &= msk0; va.q.z &= msk0; va.q.w &= msk0;
        vb.q.x &= msk1; vb.q.y &= msk1; vb.q.z &= msk1; vb.q.w &= msk1;
      }
      #pragma unroll
      for (int i = 0; i < 8; ++i)
        Bs[(nn + i) * 20 + pr] = (unsigned int)va.h[i] | ((unsigned int)vb.h[i] << 16);
    }
    __syncthreads();

    // ---- fragments + 16 WMMAs ----
    FragAB bf[4];
    #pragma unroll
    for (int nt = 0; nt < 4; ++nt) {
      const unsigned int* p = &Bs[(n0 + nt * 16 + nl) * 20 + 4 * half];
      bf[nt].q[0] = *(const uint4*)p;
      bf[nt].q[1] = *(const uint4*)(p + 8);
    }
    #pragma unroll
    for (int mt = 0; mt < 4; ++mt) {
      FragAB af;
      const unsigned short* p = &As[(m0 + mt * 16 + nl) * 40 + 8 * half];
      af.q[0] = *(const uint4*)p;
      af.q[1] = *(const uint4*)(p + 16);
      #pragma unroll
      for (int nt = 0; nt < 4; ++nt)
        acc[mt][nt] = __builtin_amdgcn_wmma_f32_16x16x32_bf16(
            false, af.v, false, bf[nt].v, (short)0, acc[mt][nt], false, false);
    }
  }

  // ---- epilogue: BN affine (+LIF spike / +identity) ----
  #pragma unroll
  for (int mt = 0; mt < 4; ++mt) {
    #pragma unroll
    for (int i = 0; i < 8; ++i) {
      const int row = mbase + m0 + mt * 16 + half * 8 + i;
      const float iv = inv[row], bs = bias[row];
      #pragma unroll
      for (int nt = 0; nt < 4; ++nt) {
        const int col = bnbase + n0 + nt * 16 + nl;
        const float y = acc[mt][nt][i] * iv + bs;
        if (MODE == 0) {
          out16[(size_t)row * BN + col] = (y >= 2.0f) ? 0x3F80 : 0;
        } else if (MODE == 1) {
          out8[(size_t)row * BN + col] = (y >= 2.0f) ? 1 : 0;
        } else {
          const size_t oi = (size_t)(col >> 10) * C * N + (size_t)row * N + (col & (N - 1));
          outf[oi] = y + xin[oi];
        }
      }
    }
  }
}

// ---- kv = sum_n k*v ; attn LIF (vth=0.5) on integer count -> (count >= 1) --
__global__ void kv_reduce(const unsigned char* __restrict__ Ks,
                          const unsigned char* __restrict__ Vs,
                          unsigned short* __restrict__ kvs) {
  int rid  = blockIdx.x * 8 + (threadIdx.x >> 5);   // rid = b*C + c
  int lane = threadIdx.x & 31;
  int b = rid / C, c = rid - b * C;
  size_t base = (size_t)c * BN + (size_t)b * N;
  const unsigned int* kp = (const unsigned int*)(Ks + base);
  const unsigned int* vp = (const unsigned int*)(Vs + base);
  unsigned int anyv = 0;
  #pragma unroll
  for (int j = 0; j < 8; ++j)
    anyv |= kp[j * 32 + lane] & vp[j * 32 + lane];
  int r = __any(anyv != 0);
  if (lane == 0) kvs[rid] = r ? 0x3F80 : 0;
}

extern "C" void kernel_launch(void* const* d_in, const int* in_sizes, int n_in,
                              void* d_out, int out_size, void* d_ws, size_t ws_size,
                              hipStream_t stream) {
  const float* x  = (const float*)d_in[0];
  const float* wq = (const float*)d_in[1];
  const float* qg = (const float*)d_in[2];
  const float* qb = (const float*)d_in[3];
  const float* qm = (const float*)d_in[4];
  const float* qv = (const float*)d_in[5];
  const float* wk = (const float*)d_in[6];
  const float* kg = (const float*)d_in[7];
  const float* kb = (const float*)d_in[8];
  const float* km = (const float*)d_in[9];
  const float* kv = (const float*)d_in[10];
  const float* wv = (const float*)d_in[11];
  const float* vg = (const float*)d_in[12];
  const float* vb = (const float*)d_in[13];
  const float* vm = (const float*)d_in[14];
  const float* vv = (const float*)d_in[15];
  const float* wp = (const float*)d_in[16];
  const float* bp = (const float*)d_in[17];
  const float* pg = (const float*)d_in[18];
  const float* pb = (const float*)d_in[19];
  const float* pm = (const float*)d_in[20];
  const float* pv = (const float*)d_in[21];

  char* ws = (char*)d_ws;
  unsigned short* W   = (unsigned short*)(ws + OFF_W);
  unsigned short* XS  = (unsigned short*)(ws + OFF_XS);
  unsigned short* Q   = (unsigned short*)(ws + OFF_Q);
  unsigned char*  KS  = (unsigned char*)(ws + OFF_KS);
  unsigned char*  VS  = (unsigned char*)(ws + OFF_VS);
  unsigned short* KVS = (unsigned short*)(ws + OFF_KVS);
  float*          P   = (float*)(ws + OFF_PAR);

  prep_params<<<3, 256, 0, stream>>>(qg, qb, qm, qv, kg, kb, km, kv,
                                     vg, vb, vm, vv, pg, pb, pm, pv, bp, P);
  convert_w<<<(4 * WELEM + 255) / 256, 256, 0, stream>>>(wq, wk, wv, wp, W);
  spike_x<<<(B * C * N + 255) / 256, 256, 0, stream>>>(x, XS);

  dim3 grid(BN / 256, C / 128);   // 64 x 6
  gemm_spike<0><<<grid, 256, 0, stream>>>(W + 0 * WELEM, XS, P + 0 * C, P + 1 * C,
                                          nullptr, nullptr, Q, nullptr, nullptr);
  gemm_spike<1><<<grid, 256, 0, stream>>>(W + 1 * WELEM, XS, P + 2 * C, P + 3 * C,
                                          nullptr, nullptr, nullptr, KS, nullptr);
  gemm_spike<1><<<grid, 256, 0, stream>>>(W + 2 * WELEM, XS, P + 4 * C, P + 5 * C,
                                          nullptr, nullptr, nullptr, VS, nullptr);
  kv_reduce<<<(B * C) / 8, 256, 0, stream>>>(KS, VS, KVS);
  gemm_spike<2><<<grid, 256, 0, stream>>>(W + 3 * WELEM, Q, P + 6 * C, P + 7 * C,
                                          KVS, x, nullptr, nullptr, (float*)d_out);
}
